// FastAttention_12171937317200
// MI455X (gfx1250) — matrix operand order repaired and checked by reading, not compiled, not run on hardware
//
#include <hip/hip_runtime.h>
#include <math.h>

// ---------------- problem constants ----------------
#define BH_   32            // B*H
#define N_    4096
#define D_    64
#define M_    266           // real feature count
#define MP_   288           // padded to 18 tiles of 16 (and divisible by 32 for WMMA K)
#define EP_   80            // 64 v-cols + 1 ones-col (denominator) + 15 pad
#define MT_   18            // MP_/16
#define ET_   5             // EP_/16
#define NORMALIZER 0.35355339059327379f   // 64^-0.25
#define RATIO      0.061313908629f        // 266^-0.5
#define EPS_       1e-4f
#define NEG_BIG    (-3.0e38f)

typedef _Float16 v16h  __attribute__((ext_vector_type(16)));
typedef _Float16 half8 __attribute__((ext_vector_type(8)));
typedef _Float16 half2 __attribute__((ext_vector_type(2)));
typedef float    v8f   __attribute__((ext_vector_type(8)));

// ---------------- WMMA helpers (gfx1250, wave32) ----------------
// A fragment (16x32 f16, M x K). base points at row 0 of the 16-row tile,
// ld = row stride in halves. Lane l: M = l&15; halves 0..15 -> K = kb + 16*(l>>4) + 0..15.
// Two contiguous 16B loads.
__device__ inline v16h load_a_frag(const _Float16* base, int ld, int kb, int lane) {
    const _Float16* p = base + (size_t)(lane & 15) * ld + kb + 16 * (lane >> 4);
    half8 lo = *(const half8*)(p);
    half8 hi = *(const half8*)(p + 8);
    v16h out;
#pragma unroll
    for (int i = 0; i < 8; ++i) { out[i] = lo[i]; out[i + 8] = hi[i]; }
    return out;
}

// B fragment (32x16 f16, K x N), stored N-major: base[col][k], ld = col stride in halves.
// Lane l: N = col0 + (l&15); K = kb + 16*(l>>4) + h, h=0..15. One contiguous 32B load.
__device__ inline v16h load_b_frag(const _Float16* base, int ld, int col0, int kb, int lane) {
    const _Float16* p = base + (size_t)(col0 + (lane & 15)) * ld + kb + 16 * (lane >> 4);
    return *(const v16h*)p;
}

__device__ inline v8f wmma_f32_f16(v16h a, v16h b, v8f c) {
    return __builtin_amdgcn_wmma_f32_16x16x32_f16(false, a, false, b, (short)0, c, false, false);
}

// ordered-uint encoding so atomicMax on unsigned == float max
__device__ inline unsigned ford(float f) {
    unsigned u = __float_as_uint(f);
    return (u & 0x80000000u) ? ~u : (u | 0x80000000u);
}
__device__ inline float ford_dec(unsigned e) {
    return (e & 0x80000000u) ? __uint_as_float(e & 0x7fffffffu) : __uint_as_float(~e);
}

// ---------------- P1: scale+convert q,k to f16; per-row diag = 0.5*sum(x^2) ----------------
// grid (BH*N, 2) x block 32.  blockIdx.y: 0 = q, 1 = k.
__global__ void p1_convert_qk(const float* __restrict__ q, const float* __restrict__ k,
                              _Float16* __restrict__ qh, _Float16* __restrict__ kh,
                              float* __restrict__ dq, float* __restrict__ dk) {
    const int lane = threadIdx.x;
    const size_t row = blockIdx.x;
    const bool isK = (blockIdx.y != 0);
    const float* src = isK ? k : q;
    _Float16* dst = isK ? kh : qh;
    float* dd = isK ? dk : dq;

    float2 v2 = ((const float2*)(src + row * D_))[lane];
    float x0 = v2.x * NORMALIZER, x1 = v2.y * NORMALIZER;
    half2 hv; hv[0] = (_Float16)x0; hv[1] = (_Float16)x1;
    ((half2*)(dst + row * D_))[lane] = hv;

    float ss = x0 * x0 + x1 * x1;
#pragma unroll
    for (int s = 1; s < 32; s <<= 1) ss += __shfl_xor(ss, s, 32);
    if (lane == 0) dd[row] = 0.5f * ss;
}

// ---------------- P2: v -> v_ext transposed e-major [bh][EP_][N_] f16 ----------------
__global__ void p2_build_vt(const float* __restrict__ v, _Float16* __restrict__ vt) {
    size_t idx = (size_t)blockIdx.x * 256 + threadIdx.x;   // [bh][e][n]
    int n = (int)(idx % N_);
    size_t t = idx / N_;
    int e = (int)(t % EP_);
    int bh = (int)(t / EP_);
    float val;
    if (e < D_)      val = v[((size_t)bh * N_ + n) * D_ + e];
    else if (e == D_) val = 1.0f;     // ones column -> k_cumsum / denominator
    else              val = 0.0f;
    vt[idx] = (_Float16)val;
}

// ---------------- P3: projection -> f16 padded [MP_][64]; init kmax slot ----------------
__global__ void p3_build_proj(const float* __restrict__ proj, _Float16* __restrict__ ph,
                              unsigned* __restrict__ kmax_enc) {
    int idx = blockIdx.x * 256 + threadIdx.x;
    if (idx < MP_ * D_) {
        int m = idx / D_;
        ph[idx] = (m < M_) ? (_Float16)proj[idx] : (_Float16)0.0f;
    }
    if (blockIdx.x == 0 && threadIdx.x == 0) *kmax_enc = 0x007FFFFFu;  // ford(-inf)
}

// ---------------- K1: global max of k_dash (WMMA GEMM + reduce + atomicMax) ----------------
// grid (N/16, BH), block 32.  All 36 WMMAs issue back-to-back (acc materialized),
// WMMA->VALU hazard paid once at the end instead of per tile.
__global__ void k1_kmax(const _Float16* __restrict__ kh, const _Float16* __restrict__ ph,
                        unsigned* __restrict__ kmax_enc) {
    const int lane = threadIdx.x;
    const int n0 = blockIdx.x * 16;
    const int bh = blockIdx.y;
    const _Float16* A = kh + ((size_t)bh * N_ + n0) * D_;

    v16h a0 = load_a_frag(A, D_, 0, lane);
    v16h a1 = load_a_frag(A, D_, 32, lane);

    v8f acc[MT_];
#pragma unroll
    for (int t = 0; t < MT_; ++t) {
        v16h b0 = load_b_frag(ph, D_, t * 16, 0, lane);
        v16h b1 = load_b_frag(ph, D_, t * 16, 32, lane);
        v8f c = {};
        c = wmma_f32_f16(a0, b0, c);
        c = wmma_f32_f16(a1, b1, c);
        acc[t] = c;
    }

    float m = NEG_BIG;
#pragma unroll
    for (int t = 0; t < MT_; ++t) {
        int mcol = t * 16 + (lane & 15);
        float valid = (mcol < M_) ? 0.0f : NEG_BIG;
#pragma unroll
        for (int g = 0; g < 8; ++g) m = fmaxf(m, acc[t][g] + valid);
    }
#pragma unroll
    for (int s = 1; s < 32; s <<= 1) m = fmaxf(m, __shfl_xor(m, s, 32));
    if (lane == 0) atomicMax(kmax_enc, ford(m));
}

// ---------------- K2: q' and k' (softmax kernel features) ----------------
// grid (N/16, BH, 2) block 32.  z: 0 = q (row max stab, n-major out), 1 = k (global stab, m-major out)
__global__ void k2_prime(const _Float16* __restrict__ qh, const _Float16* __restrict__ kh,
                         const float* __restrict__ dq, const float* __restrict__ dk,
                         const _Float16* __restrict__ ph, const unsigned* __restrict__ kmax_enc,
                         _Float16* __restrict__ qp, _Float16* __restrict__ kpt) {
    __shared__ _Float16 lbuf[16 * MP_];   // 9 KB repack tile (q branch only)

    const int lane = threadIdx.x;
    const int n0 = blockIdx.x * 16;
    const int bh = blockIdx.y;
    const bool isK = (blockIdx.z != 0);
    const size_t R0 = (size_t)bh * N_ + n0;
    const _Float16* src = isK ? kh : qh;
    const float* dg = isK ? dk : dq;

    v16h a0 = load_a_frag(src + R0 * D_, D_, 0, lane);
    v16h a1 = load_a_frag(src + R0 * D_, D_, 32, lane);

    v8f acc[MT_];
#pragma unroll
    for (int t = 0; t < MT_; ++t) {
        v16h b0 = load_b_frag(ph, D_, t * 16, 0, lane);
        v16h b1 = load_b_frag(ph, D_, t * 16, 32, lane);
        v8f c = {};
        c = wmma_f32_f16(a0, b0, c);
        c = wmma_f32_f16(a1, b1, c);
        acc[t] = c;
    }

    float dgl[8];
#pragma unroll
    for (int g = 0; g < 8; ++g) dgl[g] = dg[R0 + g + 8 * (lane >> 4)];

    float rm[8];
    if (!isK) {
#pragma unroll
        for (int g = 0; g < 8; ++g) {
            float m = NEG_BIG;
#pragma unroll
            for (int t = 0; t < MT_; ++t) {
                int mcol = t * 16 + (lane & 15);
                m = fmaxf(m, (mcol < M_) ? acc[t][g] : NEG_BIG);
            }
#pragma unroll
            for (int s = 1; s < 16; s <<= 1) m = fmaxf(m, __shfl_xor(m, s, 32));  // within 16-lane col groups
            rm[g] = m;
        }
    } else {
        float km = ford_dec(*kmax_enc);
#pragma unroll
        for (int g = 0; g < 8; ++g) rm[g] = km;
    }

    if (!isK) {
        // q': scatter f16 into LDS tile [16 rows][MP_], then stream out packed 16B stores.
#pragma unroll
        for (int t = 0; t < MT_; ++t) {
            int mcol = t * 16 + (lane & 15);
#pragma unroll
            for (int g = 0; g < 8; ++g) {
                float val = (mcol < M_)
                    ? RATIO * (__expf(acc[t][g] - dgl[g] - rm[g]) + EPS_) : 0.0f;
                lbuf[(g + 8 * (lane >> 4)) * MP_ + mcol] = (_Float16)val;
            }
        }
        __syncthreads();   // single-wave WG: compiler memory fence + dscnt ordering
        // 16 rows * 36 half8-chunks = 576 units; 18 per lane
#pragma unroll
        for (int c = 0; c < 18; ++c) {
            int u = c * 32 + lane;
            int row = u / 36;
            int moff = (u % 36) * 8;
            half8 pk = *(const half8*)(lbuf + row * MP_ + moff);
            *(half8*)(qp + (R0 + row) * MP_ + moff) = pk;
        }
    } else {
        // k': m-major [bh][MP_][N_] — lane holds 8 consecutive n for fixed m -> one 16B store
        _Float16* kb_ = kpt + (size_t)bh * MP_ * N_;
#pragma unroll
        for (int t = 0; t < MT_; ++t) {
            int mcol = t * 16 + (lane & 15);
            half8 pk;
#pragma unroll
            for (int g = 0; g < 8; ++g) {
                float val = (mcol < M_)
                    ? RATIO * (__expf(acc[t][g] - dgl[g] - rm[g]) + EPS_) : 0.0f;
                pk[g] = (_Float16)val;
            }
            *(half8*)(kb_ + (size_t)mcol * N_ + n0 + 8 * (lane >> 4)) = pk;
        }
    }
}

// ---------------- K3: context_ext[bh][EP_][MP_] = k'^T (MP_ x N) * v_ext (N x EP_) ----------------
// grid (MT_, BH), block 256 (8 waves split N), LDS cross-wave reduce
__global__ void k3_context(const _Float16* __restrict__ kpt, const _Float16* __restrict__ vt,
                           _Float16* __restrict__ ctx) {
    const int lane = threadIdx.x & 31;
    const int wave = threadIdx.x >> 5;
    const int mt = blockIdx.x;
    const int bh = blockIdx.y;
    const _Float16* A = kpt + (size_t)bh * MP_ * N_ + (size_t)(mt * 16) * N_;  // rows=m, ld=N
    const _Float16* B = vt + (size_t)bh * EP_ * N_;                            // rows=e, ld=N

    v8f acc[ET_];
#pragma unroll
    for (int et = 0; et < ET_; ++et) { v8f z = {}; acc[et] = z; }

    for (int it = 0; it < 16; ++it) {              // 8 waves * 16 iters * 32 = 4096
        int nb = wave * 512 + it * 32;
        v16h a = load_a_frag(A, N_, nb, lane);
#pragma unroll
        for (int et = 0; et < ET_; ++et) {
            v16h b = load_b_frag(B, N_, et * 16, nb, lane);
            acc[et] = wmma_f32_f16(a, b, acc[et]);
        }
    }

    __shared__ float red[8][ET_][8][32];           // 40 KB
#pragma unroll
    for (int et = 0; et < ET_; ++et)
#pragma unroll
        for (int g = 0; g < 8; ++g) red[wave][et][g][lane] = acc[et][g];
    __syncthreads();

    _Float16* cb = ctx + (size_t)bh * EP_ * MP_;
#pragma unroll
    for (int r = 0; r < 5; ++r) {                  // 1280 outputs / 256 threads
        int idx = threadIdx.x + r * 256;
        int l2 = idx & 31, g = (idx >> 5) & 7, et = idx >> 8;
        float s = 0.0f;
#pragma unroll
        for (int w = 0; w < 8; ++w) s += red[w][et][g][l2];
        int e = et * 16 + (l2 & 15);
        int m = mt * 16 + g + 8 * (l2 >> 4);
        cb[(size_t)e * MP_ + m] = (_Float16)s;     // e-major for K4 B-fragments
    }
}

// ---------------- K4: out = (q' * context_ext) / denom ----------------
// grid (N/16, BH), block 32.  Column e=64 of the accumulator is q'.k_cumsum (denominator).
__global__ void k4_output(const _Float16* __restrict__ qp, const _Float16* __restrict__ ctx,
                          float* __restrict__ out) {
    const int lane = threadIdx.x;
    const int n0 = blockIdx.x * 16;
    const int bh = blockIdx.y;
    const _Float16* A = qp + ((size_t)bh * N_ + n0) * MP_;   // rows=n, ld=MP_
    const _Float16* B = ctx + (size_t)bh * EP_ * MP_;        // rows=e, ld=MP_

    v8f acc[ET_];
#pragma unroll
    for (int et = 0; et < ET_; ++et) { v8f z = {}; acc[et] = z; }

#pragma unroll
    for (int ks = 0; ks < MP_ / 32; ++ks) {
        int kb = ks * 32;
        v16h a = load_a_frag(A, MP_, kb, lane);
#pragma unroll
        for (int et = 0; et < ET_; ++et) {
            v16h b = load_b_frag(B, MP_, et * 16, kb, lane);
            acc[et] = wmma_f32_f16(a, b, acc[et]);
        }
    }

    // denominator lives at e=64 -> tile 4, column lane%16 == 0 -> lanes 0 and 16
    float inv[8];
#pragma unroll
    for (int g = 0; g < 8; ++g) {
        float d = __shfl(acc[4][g], lane & 16, 32);
        inv[g] = 1.0f / d;
    }

    float* ob = out + ((size_t)bh * N_ + n0) * D_;
#pragma unroll
    for (int et = 0; et < 4; ++et)
#pragma unroll
        for (int g = 0; g < 8; ++g)
            ob[(size_t)(g + 8 * (lane >> 4)) * D_ + et * 16 + (lane & 15)] = acc[et][g] * inv[g];
}

// ---------------- host ----------------
extern "C" void kernel_launch(void* const* d_in, const int* in_sizes, int n_in,
                              void* d_out, int out_size, void* d_ws, size_t ws_size,
                              hipStream_t stream) {
    const float* q = (const float*)d_in[0];
    const float* k = (const float*)d_in[1];
    const float* v = (const float*)d_in[2];
    const float* proj = (const float*)d_in[3];
    float* out = (float*)d_out;
    (void)in_sizes; (void)n_in; (void)out_size; (void)ws_size;

    const size_t BHN = (size_t)BH_ * N_;
    char* ws = (char*)d_ws;
    size_t off = 0;
    auto take = [&](size_t bytes) -> char* {
        char* p = ws + off;
        off = (off + bytes + 255) & ~(size_t)255;
        return p;
    };

    _Float16* qh   = (_Float16*)take(BHN * D_ * 2);            // scaled q, f16
    _Float16* kh   = (_Float16*)take(BHN * D_ * 2);            // scaled k, f16
    float*    dq   = (float*)take(BHN * 4);                    // 0.5*sumsq (q)
    float*    dk   = (float*)take(BHN * 4);                    // 0.5*sumsq (k)
    _Float16* vt   = (_Float16*)take((size_t)BH_ * EP_ * N_ * 2); // v_ext, e-major
    _Float16* ph   = (_Float16*)take((size_t)MP_ * D_ * 2);    // projection, f16 padded
    _Float16* qp   = (_Float16*)take(BHN * MP_ * 2);           // q', n-major
    _Float16* kpt  = (_Float16*)take((size_t)BH_ * MP_ * N_ * 2); // k', m-major
    _Float16* ctx  = (_Float16*)take((size_t)BH_ * EP_ * MP_ * 2); // context_ext, e-major
    unsigned* kmax = (unsigned*)take(256);                     // global-max slot (ordered uint)

    // prep
    p1_convert_qk<<<dim3((unsigned)BHN, 2), 32, 0, stream>>>(q, k, qh, kh, dq, dk);
    {
        size_t tot = (size_t)BH_ * EP_ * N_;
        p2_build_vt<<<(unsigned)(tot / 256), 256, 0, stream>>>(v, vt);
    }
    p3_build_proj<<<(MP_ * D_ + 255) / 256, 256, 0, stream>>>(proj, ph, kmax);

    // global stabilizer for k
    k1_kmax<<<dim3(N_ / 16, BH_), 32, 0, stream>>>(kh, ph, kmax);
    // q', k'
    k2_prime<<<dim3(N_ / 16, BH_, 2), 32, 0, stream>>>(qh, kh, dq, dk, ph, kmax, qp, kpt);
    // context (+ k_cumsum as column 64)
    k3_context<<<dim3(MT_, BH_), 256, 0, stream>>>(kpt, vt, ctx);
    // output (+ denominator from column 64)
    k4_output<<<dim3(N_ / 16, BH_), 32, 0, stream>>>(qp, ctx, out);
}
